// LinearAttention_35321811042582
// MI455X (gfx1250) — compile-verified, matrix operand order checked by reading, not run on hardware
//
#include <hip/hip_runtime.h>

// ---------------- types ----------------
typedef __attribute__((ext_vector_type(16))) __bf16 v16bf;
typedef __attribute__((ext_vector_type(8)))  __bf16 v8bf;
typedef __attribute__((ext_vector_type(4)))  __bf16 v4bf;
typedef __attribute__((ext_vector_type(8)))  float  v8f;
typedef __attribute__((ext_vector_type(4)))  int    v4i;

#define DIMD       1024
#define SEQN       4096
#define NBATCH     4
#define NHEADS     16
#define HEADD      64
#define BH         (NBATCH * NHEADS)     // 64
#define MTOT       (NBATCH * SEQN)       // 16384
#define LDS_STRIDE 40                    // halves (80B rows, 16B aligned, conflict-free)

// CDNA5 async copy path (ASYNCcnt-tracked memory->LDS), guarded so a toolchain
// without the builtin falls back to load + ds_store without failing the build.
#if defined(__gfx1250__) && __has_builtin(__builtin_amdgcn_global_load_async_to_lds_b128)
#define USE_ASYNC_COPY 1
#else
#define USE_ASYNC_COPY 0
#endif

union Frag16 { v16bf v; v8bf h[2]; };

__device__ __forceinline__ v8f wmma_bf16(const Frag16& a, const Frag16& b, v8f c) {
  return __builtin_amdgcn_wmma_f32_16x16x32_bf16(false, a.v, false, b.v,
                                                 (short)0, c, false, false);
}

#if USE_ASYNC_COPY
__device__ __forceinline__ void async_copy16(const __bf16* g, __bf16* l) {
  __builtin_amdgcn_global_load_async_to_lds_b128(
      (__attribute__((address_space(1))) v4i*)(void*)g,
      (__attribute__((address_space(3))) v4i*)(void*)l,
      0, 0);
}
__device__ __forceinline__ void async_wait0() {
#if __has_builtin(__builtin_amdgcn_s_wait_asynccnt)
  __builtin_amdgcn_s_wait_asynccnt(0);
#else
  asm volatile("s_wait_asynccnt 0" ::: "memory");
#endif
}
#endif

// ---------------- f32 -> bf16 convert ----------------
__global__ __launch_bounds__(256) void cvt4(const float4* __restrict__ src,
                                            __bf16* __restrict__ dst, int n4) {
  int i = blockIdx.x * 256 + threadIdx.x;
  if (i < n4) {
    float4 f = src[i];
    v4bf o;
    o[0] = (__bf16)f.x; o[1] = (__bf16)f.y; o[2] = (__bf16)f.z; o[3] = (__bf16)f.w;
    *(v4bf*)(dst + (size_t)i * 4) = o;
  }
}

// ---------------- shared 128x128 WMMA GEMM mainloop ----------------
// C[128x128] at (m0,n0) of A[Mx1024] (row-major) times W^T, W[Nx1024] row-major.
// 256 threads = 8 waves; wave (wm in 0..3, wn in 0..1) owns 32x64 -> acc[2][4].
__device__ __forceinline__ void gemm_tile_128x128(
    const __bf16* __restrict__ A, const __bf16* __restrict__ W,
    int m0, int n0, __bf16* sA, __bf16* sB, v8f acc[2][4]) {
  const int tid   = threadIdx.x;
  const int lane  = tid & 31;
  const int wave  = tid >> 5;
  const int wm    = wave & 3;
  const int wn    = wave >> 2;
  const int lrow  = lane & 15;
  const int lhalf = lane >> 4;

  v8f z = {0.f, 0.f, 0.f, 0.f, 0.f, 0.f, 0.f, 0.f};
#pragma unroll
  for (int i = 0; i < 2; ++i)
#pragma unroll
    for (int j = 0; j < 4; ++j) acc[i][j] = z;

  for (int k0 = 0; k0 < DIMD; k0 += 32) {
    // stage A (128x32) and W (128x32) tiles: 2x 16B chunks per thread each
#pragma unroll
    for (int i = 0; i < 2; ++i) {
      int idx = tid + i * 256;
      int row = idx >> 2;
      int kc  = (idx & 3) * 8;
#if USE_ASYNC_COPY
      async_copy16(A + (size_t)(m0 + row) * DIMD + k0 + kc,
                   sA + row * LDS_STRIDE + kc);
      async_copy16(W + (size_t)(n0 + row) * DIMD + k0 + kc,
                   sB + row * LDS_STRIDE + kc);
#else
      v8bf da = *(const v8bf*)(A + (size_t)(m0 + row) * DIMD + k0 + kc);
      *(v8bf*)(sA + row * LDS_STRIDE + kc) = da;
      v8bf dw = *(const v8bf*)(W + (size_t)(n0 + row) * DIMD + k0 + kc);
      *(v8bf*)(sB + row * LDS_STRIDE + kc) = dw;
#endif
    }
    if (k0 + 32 < DIMD) {  // prefetch next K tile (global_prefetch_b8)
      __builtin_prefetch(A + (size_t)(m0 + (tid >> 1)) * DIMD + k0 + 32, 0, 1);
      __builtin_prefetch(W + (size_t)(n0 + (tid >> 1)) * DIMD + k0 + 32, 0, 1);
    }
#if USE_ASYNC_COPY
    async_wait0();   // drain this wave's ASYNCcnt before signaling the barrier
#endif
    __syncthreads();

    Frag16 a[2], b[4];
#pragma unroll
    for (int mt = 0; mt < 2; ++mt) {
      int r  = wm * 32 + mt * 16 + lrow;
      int kb = lhalf * 8;
      a[mt].h[0] = *(const v8bf*)(sA + r * LDS_STRIDE + kb);
      a[mt].h[1] = *(const v8bf*)(sA + r * LDS_STRIDE + 16 + kb);
    }
#pragma unroll
    for (int nt = 0; nt < 4; ++nt) {
      int r  = wn * 64 + nt * 16 + lrow;
      int kb = lhalf * 8;
      b[nt].h[0] = *(const v8bf*)(sB + r * LDS_STRIDE + kb);
      b[nt].h[1] = *(const v8bf*)(sB + r * LDS_STRIDE + 16 + kb);
    }
#pragma unroll
    for (int mt = 0; mt < 2; ++mt)
#pragma unroll
      for (int nt = 0; nt < 4; ++nt)
        acc[mt][nt] = wmma_bf16(a[mt], b[nt], acc[mt][nt]);
    __syncthreads();
  }
}

// ---------------- fused QKV projection (+bias, +phi) ----------------
// mode = blockIdx.z: 0 -> Q (phi, head-major), 1 -> K (phi, transposed), 2 -> V (transposed)
__global__ __launch_bounds__(256) void qkv_gemm(
    const __bf16* __restrict__ xbf,
    const __bf16* __restrict__ Wq, const __bf16* __restrict__ Wk, const __bf16* __restrict__ Wv,
    const float* __restrict__ bq, const float* __restrict__ bk, const float* __restrict__ bv,
    __bf16* __restrict__ Qh,   // [BH][N][64]
    __bf16* __restrict__ Kt,   // [BH][64][N]
    __bf16* __restrict__ Vt) { // [BH][64][N]
  __shared__ __align__(16) __bf16 sA[128 * LDS_STRIDE];
  __shared__ __align__(16) __bf16 sB[128 * LDS_STRIDE];
  const int mode = blockIdx.z;
  const __bf16* W    = (mode == 0) ? Wq : (mode == 1) ? Wk : Wv;
  const float*  bias = (mode == 0) ? bq : (mode == 1) ? bk : bv;
  const int m0 = blockIdx.y * 128;
  const int n0 = blockIdx.x * 128;

  v8f acc[2][4];
  gemm_tile_128x128(xbf, W, m0, n0, sA, sB, acc);

  const int tid = threadIdx.x, lane = tid & 31, wave = tid >> 5;
  const int wm = wave & 3, wn = wave >> 2, lrow = lane & 15, lhalf = lane >> 4;
#pragma unroll
  for (int mt = 0; mt < 2; ++mt)
#pragma unroll
    for (int nt = 0; nt < 4; ++nt)
#pragma unroll
      for (int r = 0; r < 8; ++r) {
        int row = m0 + wm * 32 + mt * 16 + r + 8 * lhalf;  // global token index
        int col = n0 + wn * 64 + nt * 16 + lrow;           // global feature index
        float v = acc[mt][nt][r] + bias[col];
        int b = row >> 12, sr = row & (SEQN - 1);
        int h = col >> 6,  d  = col & (HEADD - 1);
        int bh = b * NHEADS + h;
        if (mode == 0) {
          v = (v > 0.f) ? (v + 1.f) : __expf(v);   // phi = elu+1
          Qh[((size_t)bh * SEQN + sr) * HEADD + d] = (__bf16)v;
        } else if (mode == 1) {
          v = (v > 0.f) ? (v + 1.f) : __expf(v);
          Kt[((size_t)bh * HEADD + d) * SEQN + sr] = (__bf16)v;
        } else {
          Vt[((size_t)bh * HEADD + d) * SEQN + sr] = (__bf16)v;
        }
      }
}

// ---------------- KV = K^T V (per head), Z = sum_n K ----------------
// 1 WG per (b,h); 8 waves each reduce a 512-token slice into a 64x64 f32 tile.
__global__ __launch_bounds__(256) void kv_kernel(
    const __bf16* __restrict__ Kt, const __bf16* __restrict__ Vt,
    __bf16* __restrict__ KVt,      // [BH][64(e)][64(d)]  (stores KV transposed)
    float* __restrict__ Z) {       // [BH][64]
  const int bh = blockIdx.x;
  const __bf16* K = Kt + (size_t)bh * HEADD * SEQN;
  const __bf16* V = Vt + (size_t)bh * HEADD * SEQN;
  __shared__ __align__(16) float skv[HEADD * HEADD];
  __shared__ float sz[HEADD];
  const int tid = threadIdx.x;
  for (int i = tid; i < HEADD * HEADD; i += 256) skv[i] = 0.f;
  if (tid < HEADD) sz[tid] = 0.f;
  __syncthreads();

  const int lane = tid & 31, wave = tid >> 5;
  const int lrow = lane & 15, lhalf = lane >> 4;
  const int nbase = wave * 512;

  v8f z = {0.f, 0.f, 0.f, 0.f, 0.f, 0.f, 0.f, 0.f};
  v8f acc[4][4];
#pragma unroll
  for (int i = 0; i < 4; ++i)
#pragma unroll
    for (int j = 0; j < 4; ++j) acc[i][j] = z;

  for (int kk = 0; kk < 512; kk += 32) {
    Frag16 a[4], b[4];
#pragma unroll
    for (int mt = 0; mt < 4; ++mt) {
      const __bf16* p = K + (size_t)(mt * 16 + lrow) * SEQN + nbase + kk + lhalf * 8;
      a[mt].h[0] = *(const v8bf*)p;
      a[mt].h[1] = *(const v8bf*)(p + 16);
    }
#pragma unroll
    for (int nt = 0; nt < 4; ++nt) {
      const __bf16* p = V + (size_t)(nt * 16 + lrow) * SEQN + nbase + kk + lhalf * 8;
      b[nt].h[0] = *(const v8bf*)p;
      b[nt].h[1] = *(const v8bf*)(p + 16);
    }
#pragma unroll
    for (int mt = 0; mt < 4; ++mt)
#pragma unroll
      for (int nt = 0; nt < 4; ++nt)
        acc[mt][nt] = wmma_bf16(a[mt], b[nt], acc[mt][nt]);
  }

  // deterministic serialized cross-wave reduction into LDS
  for (int w = 0; w < 8; ++w) {
    if (wave == w) {
#pragma unroll
      for (int mt = 0; mt < 4; ++mt)
#pragma unroll
        for (int nt = 0; nt < 4; ++nt)
#pragma unroll
          for (int r = 0; r < 8; ++r) {
            int d = mt * 16 + r + 8 * lhalf;
            int e = nt * 16 + lrow;
            skv[d * HEADD + e] += acc[mt][nt][r];
          }
    }
    __syncthreads();
  }

  // Z partials: thread (part, d) sums 1024 tokens of K row d
  {
    int d = tid & (HEADD - 1);
    int part = tid >> 6;  // 0..3
    const __bf16* p = K + (size_t)d * SEQN + part * 1024;
    float s = 0.f;
    for (int i = 0; i < 1024; i += 8) {
      v8bf v = *(const v8bf*)(p + i);
#pragma unroll
      for (int j = 0; j < 8; ++j) s += (float)v[j];
    }
    for (int g = 0; g < 4; ++g) {
      if (part == g) sz[d] += s;
      __syncthreads();
    }
  }

  for (int i = tid; i < HEADD * HEADD; i += 256) {
    int d = i >> 6, e = i & (HEADD - 1);
    KVt[(size_t)bh * HEADD * HEADD + e * HEADD + d] = (__bf16)skv[d * HEADD + e];
  }
  if (tid < HEADD) Z[bh * HEADD + tid] = sz[tid];
}

// ---------------- out = (Q @ KV) / (Q.Z + eps), written token-major ----------------
__global__ __launch_bounds__(256) void attn_kernel(
    const __bf16* __restrict__ Qh,   // [BH][N][64]
    const __bf16* __restrict__ KVt,  // [BH][64(e)][64(d)]
    const float* __restrict__ Z,     // [BH][64]
    __bf16* __restrict__ ctx) {      // [B*N][1024]
  const int bh = blockIdx.x;
  const int m0 = blockIdx.y * 128;
  const __bf16* Q  = Qh  + (size_t)bh * SEQN * HEADD;
  const __bf16* Bm = KVt + (size_t)bh * HEADD * HEADD;
  const float*  z  = Z + bh * HEADD;
  __shared__ float snorm[128];
  const int tid = threadIdx.x;
  if (tid < 128) {
    const __bf16* q = Q + (size_t)(m0 + tid) * HEADD;
    float s = 0.f;
    for (int dd = 0; dd < HEADD; dd += 8) {
      v8bf qq = *(const v8bf*)(q + dd);
#pragma unroll
      for (int j = 0; j < 8; ++j) s += (float)qq[j] * z[dd + j];
    }
    snorm[tid] = 1.f / (s + 1e-6f);
  }
  __syncthreads();

  const int lane = tid & 31, wave = tid >> 5, lrow = lane & 15, lhalf = lane >> 4;
  const int mrow = m0 + wave * 16;  // 16 rows per wave
  v8f zf = {0.f, 0.f, 0.f, 0.f, 0.f, 0.f, 0.f, 0.f};
  v8f acc[4] = {zf, zf, zf, zf};
#pragma unroll
  for (int kk = 0; kk < HEADD; kk += 32) {
    Frag16 a;
    const __bf16* p = Q + (size_t)(mrow + lrow) * HEADD + kk + lhalf * 8;
    a.h[0] = *(const v8bf*)p;
    a.h[1] = *(const v8bf*)(p + 16);
#pragma unroll
    for (int nt = 0; nt < 4; ++nt) {
      Frag16 b;
      const __bf16* pb = Bm + (size_t)(nt * 16 + lrow) * HEADD + kk + lhalf * 8;
      b.h[0] = *(const v8bf*)pb;
      b.h[1] = *(const v8bf*)(pb + 16);
      acc[nt] = wmma_bf16(a, b, acc[nt]);
    }
  }

  const int b = bh >> 4, h = bh & (NHEADS - 1);
#pragma unroll
  for (int nt = 0; nt < 4; ++nt)
#pragma unroll
    for (int r = 0; r < 8; ++r) {
      int lr  = wave * 16 + r + 8 * lhalf;  // local row in 0..127
      int row = m0 + lr;                    // sequence position
      int e   = nt * 16 + lrow;
      float v = acc[nt][r] * snorm[lr];
      ctx[((size_t)(b * SEQN + row)) * DIMD + h * HEADD + e] = (__bf16)v;
    }
}

// ---------------- final projection: out = ctx @ Wo^T + bo (f32) ----------------
__global__ __launch_bounds__(256) void out_gemm(
    const __bf16* __restrict__ ctx, const __bf16* __restrict__ Wo,
    const float* __restrict__ bo, float* __restrict__ out) {
  __shared__ __align__(16) __bf16 sA[128 * LDS_STRIDE];
  __shared__ __align__(16) __bf16 sB[128 * LDS_STRIDE];
  const int m0 = blockIdx.y * 128;
  const int n0 = blockIdx.x * 128;
  v8f acc[2][4];
  gemm_tile_128x128(ctx, Wo, m0, n0, sA, sB, acc);

  const int tid = threadIdx.x, lane = tid & 31, wave = tid >> 5;
  const int wm = wave & 3, wn = wave >> 2, lrow = lane & 15, lhalf = lane >> 4;
#pragma unroll
  for (int mt = 0; mt < 2; ++mt)
#pragma unroll
    for (int nt = 0; nt < 4; ++nt)
#pragma unroll
      for (int r = 0; r < 8; ++r) {
        int row = m0 + wm * 32 + mt * 16 + r + 8 * lhalf;
        int col = n0 + wn * 64 + nt * 16 + lrow;
        out[(size_t)row * DIMD + col] = acc[mt][nt][r] + bo[col];
      }
}

// ---------------- launch ----------------
extern "C" void kernel_launch(void* const* d_in, const int* in_sizes, int n_in,
                              void* d_out, int out_size, void* d_ws, size_t ws_size,
                              hipStream_t stream) {
  const float* x  = (const float*)d_in[0];
  const float* Wq = (const float*)d_in[1];
  const float* bq = (const float*)d_in[2];
  const float* Wk = (const float*)d_in[3];
  const float* bk = (const float*)d_in[4];
  const float* Wv = (const float*)d_in[5];
  const float* bv = (const float*)d_in[6];
  const float* Wo = (const float*)d_in[7];
  const float* bo = (const float*)d_in[8];
  float* out = (float*)d_out;

  size_t off = 0;
  char* base = (char*)d_ws;
  auto alloc = [&](size_t bytes) -> void* {
    void* p = base + off;
    off += (bytes + 255) & ~(size_t)255;
    return p;
  };
  __bf16* xbf = (__bf16*)alloc((size_t)MTOT * DIMD * 2);
  __bf16* Wqb = (__bf16*)alloc((size_t)DIMD * DIMD * 2);
  __bf16* Wkb = (__bf16*)alloc((size_t)DIMD * DIMD * 2);
  __bf16* Wvb = (__bf16*)alloc((size_t)DIMD * DIMD * 2);
  __bf16* Wob = (__bf16*)alloc((size_t)DIMD * DIMD * 2);
  __bf16* Qh  = (__bf16*)alloc((size_t)MTOT * DIMD * 2);
  __bf16* Ktb = (__bf16*)alloc((size_t)MTOT * DIMD * 2);
  __bf16* Vtb = (__bf16*)alloc((size_t)MTOT * DIMD * 2);
  __bf16* KVt = (__bf16*)alloc((size_t)BH * HEADD * HEADD * 2);
  float*  Zb  = (float*) alloc((size_t)BH * HEADD * 4);
  __bf16* ctx = (__bf16*)alloc((size_t)MTOT * DIMD * 2);

  {
    int n4 = MTOT * DIMD / 4;
    cvt4<<<n4 / 256, 256, 0, stream>>>((const float4*)x, xbf, n4);
    int w4 = DIMD * DIMD / 4;
    cvt4<<<w4 / 256, 256, 0, stream>>>((const float4*)Wq, Wqb, w4);
    cvt4<<<w4 / 256, 256, 0, stream>>>((const float4*)Wk, Wkb, w4);
    cvt4<<<w4 / 256, 256, 0, stream>>>((const float4*)Wv, Wvb, w4);
    cvt4<<<w4 / 256, 256, 0, stream>>>((const float4*)Wo, Wob, w4);
  }

  qkv_gemm<<<dim3(DIMD / 128, MTOT / 128, 3), 256, 0, stream>>>(
      xbf, Wqb, Wkb, Wvb, bq, bk, bv, Qh, Ktb, Vtb);

  kv_kernel<<<dim3(BH), 256, 0, stream>>>(Ktb, Vtb, KVt, Zb);

  attn_kernel<<<dim3(BH, SEQN / 128), 256, 0, stream>>>(Qh, KVt, Zb, ctx);

  out_gemm<<<dim3(DIMD / 128, MTOT / 128), 256, 0, stream>>>(ctx, Wob, bo, out);
}
